// DecoderBlock_12824772346410
// MI455X (gfx1250) — compile-verified, matrix operand order checked by reading, not run on hardware
//
#include <hip/hip_runtime.h>

typedef __attribute__((ext_vector_type(16))) __bf16 v16bf;
typedef __attribute__((ext_vector_type(8)))  float  v8f;

namespace {
constexpr int NB   = 4;
constexpr int KSEQ = 2048;
constexpr int DM   = 512;
constexpr int FH   = 2048;
constexpr int NH   = 8;
constexpr int DH   = 64;
constexpr int MTOT = NB * KSEQ;           // 8192 rows

// workspace layout (bytes)
constexpr size_t SZ_QKV  = (size_t)NB * NH * KSEQ * DH * 2;     // 8 MiB each
constexpr size_t OFF_Q   = 0;
constexpr size_t OFF_K   = OFF_Q   + SZ_QKV;
constexpr size_t OFF_VT  = OFF_K   + SZ_QKV;
constexpr size_t OFF_ATT = OFF_VT  + SZ_QKV;
constexpr size_t OFF_H1F = OFF_ATT + (size_t)MTOT * DM * 2;
constexpr size_t OFF_H1B = OFF_H1F + (size_t)MTOT * DM * 4;
constexpr size_t OFF_ACT = OFF_H1B + (size_t)MTOT * DM * 2;
constexpr size_t OFF_XB  = OFF_ACT + (size_t)MTOT * FH * 2;     // x in bf16
constexpr size_t OFF_WQT = OFF_XB  + (size_t)MTOT * DM * 2;     // W^T bf16
constexpr size_t OFF_WKT = OFF_WQT + (size_t)NH * DM * DH * 2;
constexpr size_t OFF_WVT = OFF_WKT + (size_t)NH * DM * DH * 2;
constexpr size_t OFF_WOT = OFF_WVT + (size_t)NH * DM * DH * 2;
constexpr size_t OFF_W1T = OFF_WOT + (size_t)DM * DM * 2;
constexpr size_t OFF_W2T = OFF_W1T + (size_t)DM * FH * 2;
}

union FragU {
  v16bf v;
  unsigned short u[16];
  unsigned int   w[8];
};

__device__ __forceinline__ unsigned short f2bf(float f) {
  unsigned int x = __builtin_bit_cast(unsigned int, f);
  x += 0x7FFFu + ((x >> 16) & 1u);          // round-to-nearest-even
  return (unsigned short)(x >> 16);
}

__device__ __forceinline__ v8f vzero8() { v8f z = {0.f,0.f,0.f,0.f,0.f,0.f,0.f,0.f}; return z; }

// kk index for fragment pair p (0..7), per CDNA5 16-bit A/B layout (ISA 7.12.2)
__device__ __forceinline__ int frag_kk(int lane, int p) {
  return ((p >> 2) << 4) | ((lane >> 4) << 3) | ((p & 3) << 1);
}

// Load a 16x32 bf16 fragment where consecutive kk are contiguous in memory.
// Works for A (row-major, lane = row) and for B when memory is [col][kk].
__device__ __forceinline__ v16bf load_frag_bf16(const unsigned short* base, int ld, int lane) {
  FragU f; int r = lane & 15;
#pragma unroll
  for (int p = 0; p < 8; ++p)
    f.w[p] = *(const unsigned int*)(base + (size_t)r * ld + frag_kk(lane, p));
  return f.v;
}

__device__ __forceinline__ v8f wmma_bf16(v16bf a, v16bf b, v8f c) {
  return __builtin_amdgcn_wmma_f32_16x16x32_bf16(false, a, false, b, (short)0, c, false, false);
}

// wave-local LDS fence: order DS store -> DS load within one wave
__device__ __forceinline__ void wave_lds_fence() {
  __builtin_amdgcn_wave_barrier();
  asm volatile("s_wait_dscnt 0" ::: "memory");
  __builtin_amdgcn_wave_barrier();
}

// Software-pipelined 16xKDIMx64 GEMM core: acc[4] += A(16xKDIM) * B(64xKDIM)^T
// A: row-major [16][KDIM] at ab (ld = ldA); B: [64][KDIM] at bb (ld = ldB).
template <int KDIM>
__device__ __forceinline__ void gemm_rowtile(const unsigned short* __restrict__ ab, int ldA,
                                             const unsigned short* __restrict__ bb, int ldB,
                                             int lane, v8f acc[4]) {
  v16bf a_cur = load_frag_bf16(ab, ldA, lane);
  v16bf b_cur[4];
#pragma unroll
  for (int u = 0; u < 4; ++u)
    b_cur[u] = load_frag_bf16(bb + (size_t)(u * 16) * ldB, ldB, lane);

  for (int k0 = 0; k0 < KDIM; k0 += 32) {
    int kn = k0 + 32;
    if (kn > KDIM - 32) kn = KDIM - 32;     // clamped prefetch (last iter reloads)
    v16bf a_nxt = load_frag_bf16(ab + kn, ldA, lane);
    v16bf b_nxt[4];
#pragma unroll
    for (int u = 0; u < 4; ++u)
      b_nxt[u] = load_frag_bf16(bb + (size_t)(u * 16) * ldB + kn, ldB, lane);
#pragma unroll
    for (int u = 0; u < 4; ++u)
      acc[u] = wmma_bf16(a_cur, b_cur[u], acc[u]);
    a_cur = a_nxt;
#pragma unroll
    for (int u = 0; u < 4; ++u) b_cur[u] = b_nxt[u];
  }
}

// ---------------------------------------------------------------------------
// Kernel 0: fp32 -> bf16 convert, optionally transposing (dst[c][r] = src[r][c]).
// blockIdx.y selects a slab of rows*cols elements (for per-head weights).
// ---------------------------------------------------------------------------
__global__ void k_cvt(const float* __restrict__ src, unsigned short* __restrict__ dst,
                      int rows, int cols, int transpose) {
  size_t slab = (size_t)rows * cols;
  const float* s = src + slab * blockIdx.y;
  unsigned short* d = dst + slab * blockIdx.y;
  size_t i = (size_t)blockIdx.x * blockDim.x + threadIdx.x;
  if (i >= slab) return;
  int r = (int)(i / cols), c = (int)(i % cols);
  unsigned short b = f2bf(s[i]);
  if (transpose) d[(size_t)c * rows + r] = b;
  else           d[i] = b;
}

// ---------------------------------------------------------------------------
// Kernel 1: per-head Q/K/V projections.  grid(64, NH, 3), block 256 (8 waves)
// wave = one 16-row tile x one head's 64 cols.  V stored transposed (N,H,Dh,K).
// ---------------------------------------------------------------------------
__global__ void k_qkv(const unsigned short* __restrict__ xb,
                      const unsigned short* __restrict__ WqT,
                      const unsigned short* __restrict__ WkT,
                      const unsigned short* __restrict__ WvT,
                      const float* __restrict__ bq,
                      const float* __restrict__ bk,
                      const float* __restrict__ bv,
                      unsigned short* __restrict__ qo,
                      unsigned short* __restrict__ ko,
                      unsigned short* __restrict__ vt) {
  int lane = threadIdx.x & 31;
  int wid  = threadIdx.x >> 5;
  int rowtile = blockIdx.x * 8 + wid;       // 0..511
  int h   = blockIdx.y;
  int sel = blockIdx.z;                     // 0=q 1=k 2=v (uniform)
  const unsigned short* WT = (sel == 0) ? WqT : (sel == 1) ? WkT : WvT;
  const float* B = (sel == 0) ? bq : (sel == 1) ? bk : bv;
  WT += (size_t)h * DH * DM;                // head slab, layout [dh][d]
  B  += h * DH;

  int m0 = rowtile * 16;
  v8f acc[4];
#pragma unroll
  for (int u = 0; u < 4; ++u) acc[u] = vzero8();

  gemm_rowtile<DM>(xb + (size_t)m0 * DM, DM, WT, DM, lane, acc);

  int hi = lane >> 4, cl = lane & 15;
#pragma unroll
  for (int u = 0; u < 4; ++u) {
    float bias = B[u * 16 + cl];
#pragma unroll
    for (int r = 0; r < 8; ++r) {
      int m  = m0 + r + hi * 8;
      int nn = m >> 11;                     // m / KSEQ
      int kr = m & (KSEQ - 1);
      unsigned short b16 = f2bf(acc[u][r] + bias);
      int col = u * 16 + cl;
      if (sel == 2)
        vt[(((size_t)nn * NH + h) * DH + col) * KSEQ + kr] = b16;
      else {
        unsigned short* o = (sel == 0) ? qo : ko;
        o[(((size_t)nn * NH + h) * KSEQ + kr) * DH + col] = b16;
      }
    }
  }
}

// ---------------------------------------------------------------------------
// Kernel 2: causal flash attention.  grid(512), block 256; wave = one (n,h,16q)
// tile; keys streamed 32 at a time; online softmax; P transposed via LDS.
// K-fragments double-buffered across key blocks; V-fragments issued before the
// score WMMAs so they overlap softmax.
// ---------------------------------------------------------------------------
__global__ void k_attn(const unsigned short* __restrict__ q,
                       const unsigned short* __restrict__ kp,
                       const unsigned short* __restrict__ vt,
                       unsigned short* __restrict__ att) {
  __shared__ __align__(16) unsigned short pbuf[8 * 512];
  int lane = threadIdx.x & 31;
  int wid  = threadIdx.x >> 5;
  int tix = blockIdx.x * 8 + wid;           // 0..4095
  int qt = tix & 127;
  int h  = (tix >> 7) & 7;
  int n  = tix >> 10;
  int q0 = qt * 16;
  int hi = lane >> 4, cl = lane & 15;

  const unsigned short* qb  = q  + (((size_t)n * NH + h) * KSEQ + q0) * DH;
  const unsigned short* kbp = kp + (((size_t)n * NH + h) * KSEQ) * DH;
  const unsigned short* vbp = vt + (((size_t)n * NH + h) * DH) * KSEQ;

  v16bf qA0 = load_frag_bf16(qb,      DH, lane);   // Dh 0..31
  v16bf qA1 = load_frag_bf16(qb + 32, DH, lane);   // Dh 32..63

  float mr[8], lr[8];
#pragma unroll
  for (int r = 0; r < 8; ++r) { mr[r] = -3.0e38f; lr[r] = 0.f; }
  v8f O[4];
#pragma unroll
  for (int u = 0; u < 4; ++u) O[u] = vzero8();

  unsigned short* pw = pbuf + wid * 512;
  int niter = (q0 + 47) >> 5;               // causal: keys <= q0+15
  int kblast = (niter - 1) * 32;

  // preload K fragments for key block 0
  v16bf kf[4];
  kf[0] = load_frag_bf16(kbp,           DH, lane);
  kf[1] = load_frag_bf16(kbp + 32,      DH, lane);
  kf[2] = load_frag_bf16(kbp + 16 * DH,      DH, lane);
  kf[3] = load_frag_bf16(kbp + 16 * DH + 32, DH, lane);

  for (int it = 0; it < niter; ++it) {
    int kb = it * 32;
    // V fragments for this block: issue early, consumed only after softmax
    v16bf vf[4];
#pragma unroll
    for (int u = 0; u < 4; ++u)
      vf[u] = load_frag_bf16(vbp + (size_t)(u * 16) * KSEQ + kb, KSEQ, lane);
    // prefetch next block's K fragments (clamped on last iteration)
    int kb2 = kb + 32;
    if (kb2 > kblast) kb2 = kblast;
    v16bf kn[4];
    kn[0] = load_frag_bf16(kbp + (size_t)kb2 * DH,                DH, lane);
    kn[1] = load_frag_bf16(kbp + (size_t)kb2 * DH + 32,           DH, lane);
    kn[2] = load_frag_bf16(kbp + (size_t)(kb2 + 16) * DH,         DH, lane);
    kn[3] = load_frag_bf16(kbp + (size_t)(kb2 + 16) * DH + 32,    DH, lane);

    v8f S0 = vzero8(), S1 = vzero8();
    S0 = wmma_bf16(qA0, kf[0], S0);
    S0 = wmma_bf16(qA1, kf[1], S0);
    S1 = wmma_bf16(qA0, kf[2], S1);
    S1 = wmma_bf16(qA1, kf[3], S1);

    float alpha[8];
#pragma unroll
    for (int r = 0; r < 8; ++r) {
      int row = q0 + r + hi * 8;
      int c0 = kb + cl, c1 = kb + 16 + cl;
      float s0 = (c0 <= row) ? S0[r] * 0.125f : -1.0e9f;
      float s1 = (c1 <= row) ? S1[r] * 0.125f : -1.0e9f;
      float mx = fmaxf(s0, s1);
      mx = fmaxf(mx, __shfl_xor(mx, 1));
      mx = fmaxf(mx, __shfl_xor(mx, 2));
      mx = fmaxf(mx, __shfl_xor(mx, 4));
      mx = fmaxf(mx, __shfl_xor(mx, 8));
      float mn = fmaxf(mr[r], mx);
      float al = __expf(mr[r] - mn);
      float p0 = __expf(s0 - mn);
      float p1 = __expf(s1 - mn);
      float ps = p0 + p1;
      ps += __shfl_xor(ps, 1);
      ps += __shfl_xor(ps, 2);
      ps += __shfl_xor(ps, 4);
      ps += __shfl_xor(ps, 8);
      lr[r] = lr[r] * al + ps;
      mr[r] = mn;
      alpha[r] = al;
      pw[(r + hi * 8) * 32 + cl]      = f2bf(p0);   // P tile 16x32 in LDS
      pw[(r + hi * 8) * 32 + 16 + cl] = f2bf(p1);
    }
    wave_lds_fence();
    v16bf pA = load_frag_bf16(pw, 32, lane);        // P as A-fragment
#pragma unroll
    for (int u = 0; u < 4; ++u) {
#pragma unroll
      for (int r = 0; r < 8; ++r) O[u][r] *= alpha[r];
      O[u] = wmma_bf16(pA, vf[u], O[u]);
    }
    wave_lds_fence();
#pragma unroll
    for (int u = 0; u < 4; ++u) kf[u] = kn[u];
  }

#pragma unroll
  for (int r = 0; r < 8; ++r) {
    float inv = 1.0f / lr[r];
    int m = q0 + r + hi * 8;
#pragma unroll
    for (int u = 0; u < 4; ++u)
      att[((size_t)n * KSEQ + m) * DM + h * DH + u * 16 + cl] = f2bf(O[u][r] * inv);
  }
}

// ---------------------------------------------------------------------------
// Kernel 3: att @ Wo + bo + x, LayerNorm(g1,be1) -> h1 (fp32 + bf16).
// grid(512), block 256: 8 waves cover the full 512-col strip of 16 rows.
// ---------------------------------------------------------------------------
__global__ void k_wo_ln(const unsigned short* __restrict__ att,
                        const unsigned short* __restrict__ WoT,
                        const float* __restrict__ bo,
                        const float* __restrict__ x,
                        const float* __restrict__ g1, const float* __restrict__ be1,
                        float* __restrict__ h1f, unsigned short* __restrict__ h1b) {
  __shared__ float ssum[8][16], ssq[8][16], mrow[16], vrow[16];
  int lane = threadIdx.x & 31, wid = threadIdx.x >> 5;
  int m0 = blockIdx.x * 16;
  int n0 = wid * 64;
  int hi = lane >> 4, cl = lane & 15;

  v8f acc[4];
#pragma unroll
  for (int u = 0; u < 4; ++u) acc[u] = vzero8();

  gemm_rowtile<DM>(att + (size_t)m0 * DM, DM, WoT + (size_t)n0 * DM, DM, lane, acc);

  float y[4][8];
#pragma unroll
  for (int u = 0; u < 4; ++u) {
    int col = n0 + u * 16 + cl;
    float bias = bo[col];
#pragma unroll
    for (int r = 0; r < 8; ++r) {
      int m = m0 + r + hi * 8;
      y[u][r] = acc[u][r] + bias + x[(size_t)m * DM + col];
    }
  }
#pragma unroll
  for (int r = 0; r < 8; ++r) {
    float s  = y[0][r] + y[1][r] + y[2][r] + y[3][r];
    float q2 = y[0][r]*y[0][r] + y[1][r]*y[1][r] + y[2][r]*y[2][r] + y[3][r]*y[3][r];
    s  += __shfl_xor(s, 1);  s  += __shfl_xor(s, 2);  s  += __shfl_xor(s, 4);  s  += __shfl_xor(s, 8);
    q2 += __shfl_xor(q2, 1); q2 += __shfl_xor(q2, 2); q2 += __shfl_xor(q2, 4); q2 += __shfl_xor(q2, 8);
    if (cl == 0) { ssum[wid][r + hi * 8] = s; ssq[wid][r + hi * 8] = q2; }
  }
  __syncthreads();
  if (threadIdx.x < 16) {
    float s = 0.f, q2 = 0.f;
    for (int w = 0; w < 8; ++w) { s += ssum[w][threadIdx.x]; q2 += ssq[w][threadIdx.x]; }
    float mean = s * (1.0f / DM);
    float var  = q2 * (1.0f / DM) - mean * mean;
    mrow[threadIdx.x] = mean;
    vrow[threadIdx.x] = rsqrtf(var + 1e-10f);
  }
  __syncthreads();
#pragma unroll
  for (int r = 0; r < 8; ++r) {
    int row = r + hi * 8;
    int m = m0 + row;
    float mean = mrow[row], rstd = vrow[row];
#pragma unroll
    for (int u = 0; u < 4; ++u) {
      int col = n0 + u * 16 + cl;
      float o = g1[col] * (y[u][r] - mean) * rstd + be1[col];
      h1f[(size_t)m * DM + col] = o;
      h1b[(size_t)m * DM + col] = f2bf(o);
    }
  }
}

// ---------------------------------------------------------------------------
// Kernel 4: relu(h1 @ W1 + b1) -> bf16 activations.  grid(64, 32), block 256.
// ---------------------------------------------------------------------------
__global__ void k_ffn1(const unsigned short* __restrict__ h1b,
                       const unsigned short* __restrict__ W1T,
                       const float* __restrict__ b1,
                       unsigned short* __restrict__ act) {
  int lane = threadIdx.x & 31, wid = threadIdx.x >> 5;
  int m0 = (blockIdx.x * 8 + wid) * 16;
  int n0 = blockIdx.y * 64;
  v8f acc[4];
#pragma unroll
  for (int u = 0; u < 4; ++u) acc[u] = vzero8();

  gemm_rowtile<DM>(h1b + (size_t)m0 * DM, DM, W1T + (size_t)n0 * DM, DM, lane, acc);

  int hi = lane >> 4, cl = lane & 15;
#pragma unroll
  for (int u = 0; u < 4; ++u) {
    int col = n0 + u * 16 + cl;
    float bias = b1[col];
#pragma unroll
    for (int r = 0; r < 8; ++r) {
      int m = m0 + r + hi * 8;
      act[(size_t)m * FH + col] = f2bf(fmaxf(acc[u][r] + bias, 0.0f));
    }
  }
}

// ---------------------------------------------------------------------------
// Kernel 5: act @ W2 + b2 + h1, LayerNorm(g2,be2) -> out fp32.  grid(512).
// ---------------------------------------------------------------------------
__global__ void k_ffn2_ln(const unsigned short* __restrict__ act,
                          const unsigned short* __restrict__ W2T,
                          const float* __restrict__ b2,
                          const float* __restrict__ h1f,
                          const float* __restrict__ g2, const float* __restrict__ be2,
                          float* __restrict__ out) {
  __shared__ float ssum[8][16], ssq[8][16], mrow[16], vrow[16];
  int lane = threadIdx.x & 31, wid = threadIdx.x >> 5;
  int m0 = blockIdx.x * 16;
  int n0 = wid * 64;
  int hi = lane >> 4, cl = lane & 15;

  v8f acc[4];
#pragma unroll
  for (int u = 0; u < 4; ++u) acc[u] = vzero8();

  gemm_rowtile<FH>(act + (size_t)m0 * FH, FH, W2T + (size_t)n0 * FH, FH, lane, acc);

  float y[4][8];
#pragma unroll
  for (int u = 0; u < 4; ++u) {
    int col = n0 + u * 16 + cl;
    float bias = b2[col];
#pragma unroll
    for (int r = 0; r < 8; ++r) {
      int m = m0 + r + hi * 8;
      y[u][r] = acc[u][r] + bias + h1f[(size_t)m * DM + col];
    }
  }
#pragma unroll
  for (int r = 0; r < 8; ++r) {
    float s  = y[0][r] + y[1][r] + y[2][r] + y[3][r];
    float q2 = y[0][r]*y[0][r] + y[1][r]*y[1][r] + y[2][r]*y[2][r] + y[3][r]*y[3][r];
    s  += __shfl_xor(s, 1);  s  += __shfl_xor(s, 2);  s  += __shfl_xor(s, 4);  s  += __shfl_xor(s, 8);
    q2 += __shfl_xor(q2, 1); q2 += __shfl_xor(q2, 2); q2 += __shfl_xor(q2, 4); q2 += __shfl_xor(q2, 8);
    if (cl == 0) { ssum[wid][r + hi * 8] = s; ssq[wid][r + hi * 8] = q2; }
  }
  __syncthreads();
  if (threadIdx.x < 16) {
    float s = 0.f, q2 = 0.f;
    for (int w = 0; w < 8; ++w) { s += ssum[w][threadIdx.x]; q2 += ssq[w][threadIdx.x]; }
    float mean = s * (1.0f / DM);
    float var  = q2 * (1.0f / DM) - mean * mean;
    mrow[threadIdx.x] = mean;
    vrow[threadIdx.x] = rsqrtf(var + 1e-10f);
  }
  __syncthreads();
#pragma unroll
  for (int r = 0; r < 8; ++r) {
    int row = r + hi * 8;
    int m = m0 + row;
    float mean = mrow[row], rstd = vrow[row];
#pragma unroll
    for (int u = 0; u < 4; ++u) {
      int col = n0 + u * 16 + cl;
      out[(size_t)m * DM + col] = g2[col] * (y[u][r] - mean) * rstd + be2[col];
    }
  }
}

extern "C" void kernel_launch(void* const* d_in, const int* in_sizes, int n_in,
                              void* d_out, int out_size, void* d_ws, size_t ws_size,
                              hipStream_t stream) {
  (void)in_sizes; (void)n_in; (void)out_size; (void)ws_size;
  const float* x   = (const float*)d_in[0];
  const float* Wq  = (const float*)d_in[1];
  const float* bq  = (const float*)d_in[2];
  const float* Wk  = (const float*)d_in[3];
  const float* bk  = (const float*)d_in[4];
  const float* Wv  = (const float*)d_in[5];
  const float* bv  = (const float*)d_in[6];
  const float* Wo  = (const float*)d_in[7];
  const float* bo  = (const float*)d_in[8];
  const float* W1  = (const float*)d_in[9];
  const float* b1  = (const float*)d_in[10];
  const float* W2  = (const float*)d_in[11];
  const float* b2  = (const float*)d_in[12];
  const float* g1  = (const float*)d_in[13];
  const float* be1 = (const float*)d_in[14];
  const float* g2  = (const float*)d_in[15];
  const float* be2 = (const float*)d_in[16];
  // d_in[17] = causal mask, implicit in kernel 2
  float* out = (float*)d_out;

  char* ws = (char*)d_ws;
  unsigned short* qb   = (unsigned short*)(ws + OFF_Q);
  unsigned short* kb   = (unsigned short*)(ws + OFF_K);
  unsigned short* vtb  = (unsigned short*)(ws + OFF_VT);
  unsigned short* attb = (unsigned short*)(ws + OFF_ATT);
  float*          h1f  = (float*)(ws + OFF_H1F);
  unsigned short* h1bb = (unsigned short*)(ws + OFF_H1B);
  unsigned short* actb = (unsigned short*)(ws + OFF_ACT);
  unsigned short* xb   = (unsigned short*)(ws + OFF_XB);
  unsigned short* WqT  = (unsigned short*)(ws + OFF_WQT);
  unsigned short* WkT  = (unsigned short*)(ws + OFF_WKT);
  unsigned short* WvT  = (unsigned short*)(ws + OFF_WVT);
  unsigned short* WoT  = (unsigned short*)(ws + OFF_WOT);
  unsigned short* W1T  = (unsigned short*)(ws + OFF_W1T);
  unsigned short* W2T  = (unsigned short*)(ws + OFF_W2T);

  // one-time fp32 -> bf16 conversions (weights transposed to [n][k])
  k_cvt<<<dim3((MTOT * DM + 255) / 256, 1),  256, 0, stream>>>(x,  xb,  MTOT, DM, 0);
  k_cvt<<<dim3((DM * DH + 255) / 256, NH),   256, 0, stream>>>(Wq, WqT, DM, DH, 1);
  k_cvt<<<dim3((DM * DH + 255) / 256, NH),   256, 0, stream>>>(Wk, WkT, DM, DH, 1);
  k_cvt<<<dim3((DM * DH + 255) / 256, NH),   256, 0, stream>>>(Wv, WvT, DM, DH, 1);
  k_cvt<<<dim3((DM * DM + 255) / 256, 1),    256, 0, stream>>>(Wo, WoT, DM, DM, 1);
  k_cvt<<<dim3((DM * FH + 255) / 256, 1),    256, 0, stream>>>(W1, W1T, DM, FH, 1);
  k_cvt<<<dim3((FH * DM + 255) / 256, 1),    256, 0, stream>>>(W2, W2T, FH, DM, 1);

  k_qkv    <<<dim3(64, NH, 3), 256, 0, stream>>>(xb, WqT, WkT, WvT, bq, bk, bv, qb, kb, vtb);
  k_attn   <<<dim3(512),       256, 0, stream>>>(qb, kb, vtb, attb);
  k_wo_ln  <<<dim3(512),       256, 0, stream>>>(attb, WoT, bo, x, g1, be1, h1f, h1bb);
  k_ffn1   <<<dim3(64, 32),    256, 0, stream>>>(h1bb, W1T, b1, actb);
  k_ffn2_ln<<<dim3(512),       256, 0, stream>>>(actb, W2T, b2, h1f, g2, be2, out);
}
